// LastAggregator_28561532518860
// MI455X (gfx1250) — matrix operand order repaired and checked by reading, not compiled, not run on hardware
//
#include <hip/hip_runtime.h>
#include <math.h>

#define DIM 128
#define BB  256
#define SS  100
#define NN  16
#define ROWS (BB*SS)   // 25600

typedef __attribute__((ext_vector_type(16))) __bf16   v16bf;
typedef __attribute__((ext_vector_type(8)))  float    v8f;
typedef __attribute__((ext_vector_type(4)))  unsigned v4u;
typedef __attribute__((ext_vector_type(4)))  int      v4i;

union AFrag { v16bf v; unsigned u[8]; };
union BFrag { v16bf v; v4u q[2]; };

// ---- CDNA5 async global->LDS DMA (ASYNCcnt path), guarded so compile never breaks ----
#if defined(__has_builtin)
# if __has_builtin(__builtin_amdgcn_global_load_async_to_lds_b128) && \
     __has_builtin(__builtin_amdgcn_s_wait_asynccnt)
#  define USE_ASYNC_LDS 1
# endif
#endif
#ifndef USE_ASYNC_LDS
# define USE_ASYNC_LDS 0
#endif

__device__ __forceinline__ void stage16(const float* __restrict__ g, float* l) {
#if USE_ASYNC_LDS
  void* gg = (void*)g;   // strip const first; then addrspace+type cast
  __builtin_amdgcn_global_load_async_to_lds_b128(
      (__attribute__((address_space(1))) v4i*)gg,
      (__attribute__((address_space(3))) v4i*)l, 0, 0);
#else
  *(float4*)l = *(const float4*)g;
#endif
}
__device__ __forceinline__ void stage_wait() {
#if USE_ASYNC_LDS
  __builtin_amdgcn_s_wait_asynccnt(0);
#endif
}

__device__ __forceinline__ __bf16 f2bf(float x) {
  union { float f; unsigned u; } c; c.f = x;
  unsigned r = c.u + 0x7FFFu + ((c.u >> 16) & 1u);   // round-to-nearest-even
  union { unsigned short s; __bf16 b; } o; o.s = (unsigned short)(r >> 16);
  return o.b;
}

__device__ __forceinline__ float sigmoidf(float x) {
  return 1.f / (1.f + __expf(-x));
}

// One wave computes a 16x16 f32 tile:  A(16x128 bf16, row-major in LDS) x B(128x16, pre-packed)
// A layout (ISA 7.12.2, 16-bit A 16x32): lane<16 -> K {0..7,16..23}; lane>=16 -> K {8..15,24..31}
// B layout: lane<16 -> N=lane, K {0..7,16..23}; lane>=16 -> N=lane-16, K {8..15,24..31} (pre-baked by k_pack)
__device__ __forceinline__ v8f wmma_rowtile(const __bf16* a_lds, const __bf16* bpack_ct,
                                            int lane, v8f acc) {
  const int m    = lane & 15;
  const int half = lane >> 4;
  const unsigned* arow = (const unsigned*)(a_lds + m * DIM);   // 64 dwords/row
  #pragma unroll
  for (int kc = 0; kc < 4; ++kc) {
    AFrag A;
    const int base = kc * 16 + half * 4;
    #pragma unroll
    for (int p = 0; p < 8; ++p)
      A.u[p] = arow[base + p + ((p >= 4) ? 4 : 0)];
    BFrag Bf;
    const v4u* bp = (const v4u*)(bpack_ct + (kc * 32 + lane) * 16);
    Bf.q[0] = bp[0]; Bf.q[1] = bp[1];
    acc = __builtin_amdgcn_wmma_f32_16x16x32_bf16(false, A.v, false, Bf.v,
                                                  (short)0, acc, false, false);
  }
  return acc;
}

// ---------------- kernel 1: Kmat = k1 @ q1^T  (128x128) ----------------
__global__ void k_kmat(const float* __restrict__ q1, const float* __restrict__ k1,
                       float* __restrict__ Kmat) {
  const int idx = blockIdx.x * 256 + threadIdx.x;      // 16384 outputs
  const int dp = idx >> 7, d = idx & 127;
  const float* kr = k1 + dp * DIM;
  const float* qr = q1 + d  * DIM;
  float s = 0.f;
  #pragma unroll 8
  for (int e = 0; e < DIM; ++e) s += kr[e] * qr[e];
  Kmat[dp * DIM + d] = s;
}

// ---------------- kernel 2: pack 4 weight matrices into WMMA-B fragment order ----------------
// packs[mat][c][kc][lane][e] = bf16( W[K][n] ),  K = kc*32 + (lane>=16?8:0) + (e&7) + (e>=8?16:0),  n = c*16 + (lane&15)
__global__ void k_pack(const float* __restrict__ Kmat, const float* __restrict__ w1,
                       const float* __restrict__ w2, const float* __restrict__ lin1_w,
                       __bf16* __restrict__ packs) {
  const int idx = blockIdx.x * 256 + threadIdx.x;      // 4 * 16384
  const int mat = idx >> 14;
  const int pos = idx & 16383;
  const int e    = pos & 15;
  const int lane = (pos >> 4) & 31;
  const int kc   = (pos >> 9) & 3;
  const int c    = pos >> 11;
  const int K = kc * 32 + ((lane & 16) ? 8 : 0) + (e & 7) + ((e & 8) ? 16 : 0);
  const int n = c * 16 + (lane & 15);
  float v;
  if (mat == 0)      v = Kmat[K * DIM + n];
  else if (mat == 1) v = w1[K * DIM + n];
  else if (mat == 2) v = w2[K * DIM + n];
  else               v = lin1_w[n * DIM + K];          // lin1_w transposed
  packs[idx] = f2bf(v);
}

// ---------------- kernel 3: t = item @ Kmat  (WMMA) ----------------
__global__ void k_gemm_t(const float* __restrict__ X, const __bf16* __restrict__ bpack,
                         float* __restrict__ Y) {
  __shared__ __align__(16) float  a_f32[16 * DIM];     // async-DMA staged
  __shared__ __align__(16) __bf16 a_lds[16 * DIM];
  const int r0 = blockIdx.x * 16;
  const int tid = threadIdx.x;
  for (int i = tid; i < 512; i += 256)                 // 512 x b128 chunks
    stage16(X + (size_t)r0 * DIM + i * 4, a_f32 + i * 4);
  stage_wait();
  __syncthreads();
  for (int i = tid; i < 16 * DIM; i += 256) a_lds[i] = f2bf(a_f32[i]);
  __syncthreads();
  const int wave = tid >> 5, lane = tid & 31;
  v8f acc = {};
  acc = wmma_rowtile(a_lds, bpack + wave * 2048, lane, acc);
  const int col  = wave * 16 + (lane & 15);
  const int rowb = r0 + (lane >> 4) * 8;
  #pragma unroll
  for (int i = 0; i < 8; ++i) Y[(rowb + i) * DIM + col] = acc[i];
}

// ---------------- kernel 4: alpha + agg, single pass over adj (bandwidth-critical) ----------------
__global__ void k_alpha(const float* __restrict__ adj, const float* __restrict__ t,
                        float* __restrict__ agg) {
  const int gw   = (blockIdx.x * blockDim.x + threadIdx.x) >> 5;  // one (b,s) per wave
  const int lane = threadIdx.x & 31;
  const float4 tv = ((const float4*)(t + (size_t)gw * DIM))[lane];
  const float4* A = (const float4*)(adj + (size_t)gw * (NN * DIM));
  float4 acc = make_float4(0.f, 0.f, 0.f, 0.f);
  #pragma unroll
  for (int n = 0; n < NN; ++n) {
    const float4 av = A[n * 32 + lane];
    float p = av.x * tv.x + av.y * tv.y + av.z * tv.z + av.w * tv.w;
    p += __shfl_xor(p, 16, 32);
    p += __shfl_xor(p,  8, 32);
    p += __shfl_xor(p,  4, 32);
    p += __shfl_xor(p,  2, 32);
    p += __shfl_xor(p,  1, 32);
    const float alpha = p * 0.08838834764831845f;      // 1/sqrt(128)
    acc.x += alpha * av.x; acc.y += alpha * av.y;
    acc.z += alpha * av.z; acc.w += alpha * av.w;
  }
  ((float4*)(agg + (size_t)gw * DIM))[lane] = acc;
}

// ---------------- kernel 5: weight = sigmoid(item@w1 + agg@w2); final = lerp ----------------
__global__ void k_final(const float* __restrict__ item, const float* __restrict__ agg,
                        const __bf16* __restrict__ packW1, const __bf16* __restrict__ packW2,
                        float* __restrict__ outF) {
  __shared__ __align__(16) float fi[16 * DIM], fg[16 * DIM];   // async-DMA staged
  __shared__ __align__(16) __bf16 ai[16 * DIM], ag[16 * DIM];
  const int r0 = blockIdx.x * 16;
  const int tid = threadIdx.x;
  for (int i = tid; i < 512; i += 256) {
    stage16(item + (size_t)r0 * DIM + i * 4, fi + i * 4);
    stage16(agg  + (size_t)r0 * DIM + i * 4, fg + i * 4);
  }
  stage_wait();
  __syncthreads();
  for (int i = tid; i < 16 * DIM; i += 256) { ai[i] = f2bf(fi[i]); ag[i] = f2bf(fg[i]); }
  __syncthreads();
  const int wave = tid >> 5, lane = tid & 31;
  v8f acc = {};
  acc = wmma_rowtile(ai, packW1 + wave * 2048, lane, acc);
  acc = wmma_rowtile(ag, packW2 + wave * 2048, lane, acc);
  const int col = wave * 16 + (lane & 15);
  const int lr0 = (lane >> 4) * 8;
  #pragma unroll
  for (int i = 0; i < 8; ++i) {
    const int lr = lr0 + i;
    const float w = sigmoidf(acc[i]);
    const float f = (1.f - w) * fi[lr * DIM + col] + w * fg[lr * DIM + col];
    outF[(r0 + lr) * DIM + col] = f;
  }
}

// ---------------- kernel 6: sequence attention pooling, one block per b ----------------
__global__ void k_pool(const float* __restrict__ Fall, const float* __restrict__ lin1_b,
                       const float* __restrict__ lin2_w, const float* __restrict__ linw_w,
                       const __bf16* __restrict__ packL1, float* __restrict__ out) {
  const int b = blockIdx.x, tid = threadIdx.x;
  const float* F = Fall + (size_t)b * SS * DIM;
  __shared__ __align__(16) float  F_lds[SS * DIM];     // 50 KB: final block, async-DMA staged
  __shared__ __align__(16) __bf16 a_lds[112 * DIM];    // 28 KB, S padded to 112
  __shared__ float avg[DIM], c2[DIM], betap[8][112], beta[112], osum[2][DIM];
  for (int i = tid; i < (SS * DIM) / 4; i += 256)      // 3200 x b128 chunks
    stage16(F + i * 4, F_lds + i * 4);
  stage_wait();
  __syncthreads();
  for (int i = tid; i < 112 * DIM; i += 256)
    a_lds[i] = (i < SS * DIM) ? f2bf(F_lds[i]) : f2bf(0.f);
  if (tid < DIM) {
    float s = 0.f;
    for (int ss = 0; ss < SS; ++ss) s += F_lds[ss * DIM + tid];
    avg[tid] = s * (1.f / SS);
  }
  __syncthreads();
  if (tid < DIM) {
    float s = lin1_b[tid];
    const float* l2r = lin2_w + tid * DIM;
    #pragma unroll 8
    for (int d = 0; d < DIM; ++d) s += avg[d] * l2r[d];
    c2[tid] = s;
  }
  __syncthreads();
  const int wave = tid >> 5, lane = tid & 31;
  const int col  = wave * 16 + (lane & 15);
  const int half = lane >> 4;
  const float lw  = linw_w[col];
  const float c2c = c2[col];
  for (int mt = 0; mt < 7; ++mt) {                     // 7 M-tiles, uniform across waves
    v8f acc = {};
    acc = wmma_rowtile(a_lds + mt * 16 * DIM, packL1 + wave * 2048, lane, acc);
    #pragma unroll
    for (int i = 0; i < 8; ++i) {
      float contrib = sigmoidf(acc[i] + c2c) * lw;     // beta contribution of this column
      contrib += __shfl_xor(contrib, 1, 32);           // reduce 16 cols within lane-half
      contrib += __shfl_xor(contrib, 2, 32);
      contrib += __shfl_xor(contrib, 4, 32);
      contrib += __shfl_xor(contrib, 8, 32);
      if ((lane & 15) == 0) betap[wave][mt * 16 + half * 8 + i] = contrib;
    }
  }
  __syncthreads();
  if (tid < 112) {
    float s = 0.f;
    #pragma unroll
    for (int w = 0; w < 8; ++w) s += betap[w][tid];    // deterministic cross-wave combine
    beta[tid] = s;
  }
  __syncthreads();
  {
    const int hs = tid >> 7, d = tid & 127;
    float s = 0.f;
    for (int ss = hs * 50; ss < hs * 50 + 50; ++ss) s += beta[ss] * F_lds[ss * DIM + d];
    osum[hs][d] = s;
  }
  __syncthreads();
  if (tid < DIM) out[b * DIM + tid] = osum[0][tid] + osum[1][tid];
}

extern "C" void kernel_launch(void* const* d_in, const int* in_sizes, int n_in,
                              void* d_out, int out_size, void* d_ws, size_t ws_size,
                              hipStream_t stream) {
  (void)in_sizes; (void)n_in; (void)out_size; (void)ws_size;
  const float* item  = (const float*)d_in[0];
  const float* adj   = (const float*)d_in[1];
  const float* q1    = (const float*)d_in[2];
  const float* k1    = (const float*)d_in[3];
  const float* w1    = (const float*)d_in[4];
  const float* w2    = (const float*)d_in[5];
  const float* lin1w = (const float*)d_in[6];
  const float* lin1b = (const float*)d_in[7];
  const float* lin2w = (const float*)d_in[8];
  const float* linww = (const float*)d_in[9];
  float* out = (float*)d_out;

  char* ws = (char*)d_ws;
  float*  Kmat   = (float*) (ws + 0);           // 64 KB
  __bf16* packs  = (__bf16*)(ws + 65536);       // 4 x 32 KB
  float*  t_ws   = (float*) (ws + 196608);      // 12.5 MB
  float*  agg_ws = (float*) (ws + 13303808);    // 12.5 MB
  float*  fin_ws = t_ws;                        // t is dead after k_alpha; reuse

  k_kmat  <<<  64, 256, 0, stream>>>(q1, k1, Kmat);
  k_pack  <<< 256, 256, 0, stream>>>(Kmat, w1, w2, lin1w, packs);
  k_gemm_t<<<1600, 256, 0, stream>>>(item, packs, t_ws);
  k_alpha <<<3200, 256, 0, stream>>>(adj, t_ws, agg_ws);
  k_final <<<1600, 256, 0, stream>>>(item, agg_ws, packs + 16384, packs + 2 * 16384, fin_ws);
  k_pool  <<< 256, 256, 0, stream>>>(fin_ws, lin1b, lin2w, linww, packs + 3 * 16384, out);
}
